// HTM_50354196579134
// MI455X (gfx1250) — matrix operand order repaired and checked by reading, not compiled, not run on hardware
//
#include <hip/hip_runtime.h>
#include <hip/hip_fp16.h>
#include <math.h>

// ---------------- problem constants ----------------
#define BB    128      // batch
#define NTOK  16       // N
#define INPD  1024
#define ED    512      // E
#define MUD   512
#define MM    15       // M = N-1
#define TT    31       // T = N+M
#define LL    30       // L = N+M-1
#define P2T   (LL*LL)  // 900
#define MASKV (-9e20f)
#define SFT_EPS 1e-20f
#define STD_EPS 1e-20f

typedef __attribute__((ext_vector_type(16))) _Float16 v16h;
typedef __attribute__((ext_vector_type(8)))  _Float16 v8h;
typedef __attribute__((ext_vector_type(8)))  float    v8f;
typedef unsigned int u32x4 __attribute__((ext_vector_type(4)));
typedef int          i32x8 __attribute__((ext_vector_type(8)));
typedef int          i32x4 __attribute__((ext_vector_type(4)));

#if defined(__gfx1250__) && __has_builtin(__builtin_amdgcn_tensor_load_to_lds)
#define USE_TDM 1
#else
#define USE_TDM 0
#endif

// ---------------- tiny elementwise kernels ----------------
__global__ void k_fill(float* p, float v, int n) {
  int i = blockIdx.x * blockDim.x + threadIdx.x;
  if (i < n) p[i] = v;
}

__global__ void k_copy(float* dst, const float* src, int n) {
  int i = blockIdx.x * blockDim.x + threadIdx.x;
  if (i < n) dst[i] = src[i];
}

__global__ void k_scale_copy(float* dst, const float* src, float s, int n) {
  int i = blockIdx.x * blockDim.x + threadIdx.x;
  if (i < n) dst[i] = src[i] * s;
}

__global__ void k_cvt_f16(const float* src, _Float16* dst, int n) {
  int i = blockIdx.x * blockDim.x + threadIdx.x;
  if (i < n) dst[i] = (_Float16)src[i];
}

// W[K][N] f32 -> Wt[N][K] f16
__global__ void k_cvt_tr_f16(const float* __restrict__ W, _Float16* __restrict__ Wt,
                             int K, int Nc) {
  int i = blockIdx.x * blockDim.x + threadIdx.x;
  if (i >= K * Nc) return;
  int k = i / Nc, n = i % Nc;
  Wt[(size_t)n * K + k] = (_Float16)W[i];
}

__global__ void k_labels(float* dst) {
  int i = blockIdx.x * blockDim.x + threadIdx.x;
  if (i < BB * TT) dst[i] = ((i % TT) < MM) ? 1.0f : 0.0f;
}

// ---------------- WMMA f16 GEMM: C = act(A[M][K] @ Wt[N][K]^T + bias) ----------------
#define TBM 128
#define TBN 64
#define TBK 32

#if USE_TDM
// Issue one TDM 2D tile load: tileRows x TBK halfs, row stride = strideElems halfs.
__device__ __forceinline__ void tdm_load_tile(unsigned lds_off, const _Float16* gsrc,
                                              int strideElems, int tileRows) {
  unsigned long long ga = (unsigned long long)(uintptr_t)gsrc;
  u32x4 g0;
  g0[0] = 1u;                                               // count=1, user D#
  g0[1] = lds_off;                                          // lds_addr (bytes)
  g0[2] = (unsigned)(ga & 0xffffffffull);                   // global_addr lo
  g0[3] = (unsigned)((ga >> 32) & 0x01ffffffull) | (2u << 30); // addr hi + type=2
  i32x8 g1;
  unsigned K = (unsigned)strideElems;
  g1[0] = (int)(1u << 16);                                  // data_size = 2 bytes
  g1[1] = (int)((K & 0xffffu) << 16);                       // tensor_dim0 lo16
  g1[2] = (int)((K >> 16) & 0xffffu);                       // tensor_dim0 hi16, dim1 lo16=0
  g1[3] = (int)(0x10u | ((unsigned)TBK << 16));             // dim1 hi16 (dim1=1M), tile_dim0=32
  g1[4] = (int)((unsigned)tileRows & 0xffffu);              // tile_dim1; tile_dim2=0
  g1[5] = (int)K;                                           // tensor_dim0_stride lo32
  g1[6] = 0;                                                // stride hi16; dim1_stride lo16
  g1[7] = 0;
  i32x4 z4 = {0, 0, 0, 0};
  i32x8 z8 = {0, 0, 0, 0, 0, 0, 0, 0};
  __builtin_amdgcn_tensor_load_to_lds(g0, g1, z4, z4, z8, 0);
}
#endif

__launch_bounds__(256)
__global__ void k_gemm_f16(const _Float16* __restrict__ A,
                           const _Float16* __restrict__ Bt,   // [Nc][K]
                           const float* __restrict__ bias,    // [Nc] or null
                           float* __restrict__ C,             // [Mr][Nc] or null
                           _Float16* __restrict__ Ch,         // [Mr][Nc] f16 or null
                           int K, int Nc, int relu) {
  __shared__ _Float16 sA[2][TBM * TBK];
  __shared__ _Float16 sB[2][TBN * TBK];
  const int row0 = blockIdx.x * TBM;
  const int col0 = blockIdx.y * TBN;
  const int tid  = threadIdx.x;
  const int wave = tid >> 5;
  const int lane = tid & 31;
  const int wm = wave >> 1;   // 0..3
  const int wn = wave & 1;    // 0..1
  const int l16  = lane & 15;
  const int hsel = lane >> 4;

  v8f acc00 = {}, acc01 = {}, acc10 = {}, acc11 = {};

#if USE_TDM
  const unsigned sAoff0 = (unsigned)(uintptr_t)&sA[0][0];
  const unsigned sAoff1 = (unsigned)(uintptr_t)&sA[1][0];
  const unsigned sBoff0 = (unsigned)(uintptr_t)&sB[0][0];
  const unsigned sBoff1 = (unsigned)(uintptr_t)&sB[1][0];
  if (wave == 0) {
    tdm_load_tile(sAoff0, A + (size_t)row0 * K, K, TBM);
    tdm_load_tile(sBoff0, Bt + (size_t)col0 * K, K, TBN);
    __builtin_amdgcn_s_wait_tensorcnt(0);
  }
  __syncthreads();
#else
  const int ar = tid >> 1;          // 0..127 (A row in tile)
  const int ap = (tid & 1) * 16;    // half-offset 0 or 16
  const int bn = tid >> 2;          // 0..63  (B n-row in tile)
  const int bq = (tid & 3) * 8;     // half-offset 0,8,16,24
#endif

  const int nk = K / TBK;
  for (int ki = 0; ki < nk; ++ki) {
    const int buf = ki & 1;
#if USE_TDM
    if (wave == 0 && ki + 1 < nk) {
      const int k0n = (ki + 1) * TBK;
      tdm_load_tile(buf ? sAoff0 : sAoff1, A + (size_t)row0 * K + k0n, K, TBM);
      tdm_load_tile(buf ? sBoff0 : sBoff1, Bt + (size_t)col0 * K + k0n, K, TBN);
    }
#else
    const int k0 = ki * TBK;
    {
      const _Float16* ga = A + (size_t)(row0 + ar) * K + k0 + ap;
      *(v8h*)&sA[buf][ar * TBK + ap]     = *(const v8h*)ga;
      *(v8h*)&sA[buf][ar * TBK + ap + 8] = *(const v8h*)(ga + 8);
      const _Float16* gb = Bt + (size_t)(col0 + bn) * K + k0 + bq;
      *(v8h*)&sB[buf][bn * TBK + bq] = *(const v8h*)gb;
    }
    __syncthreads();
#endif

    #pragma unroll
    for (int i = 0; i < 2; ++i) {
      const int mo = wm * 32 + i * 16;
      const _Float16* pa = &sA[buf][(mo + l16) * TBK + hsel * 8];
      v8h a0 = *(const v8h*)pa;
      v8h a1 = *(const v8h*)(pa + 16);
      v16h av;
      #pragma unroll
      for (int t = 0; t < 8; ++t) { av[t] = a0[t]; av[t + 8] = a1[t]; }
      #pragma unroll
      for (int j = 0; j < 2; ++j) {
        const int no = wn * 32 + j * 16;
        const _Float16* pb = &sB[buf][(no + l16) * TBK + hsel * 8];
        v8h b0 = *(const v8h*)pb;
        v8h b1 = *(const v8h*)(pb + 16);
        v16h bv;
        #pragma unroll
        for (int t = 0; t < 8; ++t) { bv[t] = b0[t]; bv[t + 8] = b1[t]; }
        v8f cc = (i == 0) ? (j == 0 ? acc00 : acc01) : (j == 0 ? acc10 : acc11);
        cc = __builtin_amdgcn_wmma_f32_16x16x32_f16(false, av, false, bv,
                                                    (short)0, cc, false, false);
        if (i == 0) { if (j == 0) acc00 = cc; else acc01 = cc; }
        else        { if (j == 0) acc10 = cc; else acc11 = cc; }
      }
    }

#if USE_TDM
    if (wave == 0) __builtin_amdgcn_s_wait_tensorcnt(0);
#endif
    __syncthreads();
  }

  const int rsel = hsel * 8;
  #pragma unroll
  for (int i = 0; i < 2; ++i) {
    #pragma unroll
    for (int j = 0; j < 2; ++j) {
      v8f cc = (i == 0) ? (j == 0 ? acc00 : acc01) : (j == 0 ? acc10 : acc11);
      const int gn = col0 + wn * 32 + j * 16 + l16;
      const float bv = bias ? bias[gn] : 0.0f;
      #pragma unroll
      for (int r = 0; r < 8; ++r) {
        const int gm = row0 + wm * 32 + i * 16 + rsel + r;
        float v = cc[r] + bv;
        if (relu) v = v > 0.0f ? v : 0.0f;
        if (C)  C[(size_t)gm * Nc + gn]  = v;
        if (Ch) Ch[(size_t)gm * Nc + gn] = (_Float16)v;
      }
    }
  }
}

// ---------------- head kernels (Nc = 1 or 2) ----------------
__global__ void k_rowdot1(const _Float16* __restrict__ H, const float* __restrict__ w,
                          const float* __restrict__ b, float* __restrict__ out, int K) {
  const int row = blockIdx.x;
  const int tid = threadIdx.x;
  float s = 0.0f;
  for (int k = tid; k < K; k += 128) s += (float)H[(size_t)row * K + k] * w[k];
  __shared__ float red[128];
  red[tid] = s; __syncthreads();
  for (int o = 64; o > 0; o >>= 1) { if (tid < o) red[tid] += red[tid + o]; __syncthreads(); }
  if (tid == 0) out[row] = red[0] + b[0];
}

// mode 0: rows = BB (b), write clf_pred[b][it];  mode 1: rows = BB*NTOK, write clf_pred[b][MM + (NTOK-1-j)]
__global__ void k_clf_head(const _Float16* __restrict__ H, const float* __restrict__ w, // [K][2]
                           const float* __restrict__ b2, float* __restrict__ clfpred,
                           int K, int mode, int it) {
  const int row = blockIdx.x;
  const int tid = threadIdx.x;
  float s0 = 0.0f, s1 = 0.0f;
  for (int k = tid; k < K; k += 128) {
    float h = (float)H[(size_t)row * K + k];
    s0 += h * w[k * 2];
    s1 += h * w[k * 2 + 1];
  }
  __shared__ float r0[128], r1[128];
  r0[tid] = s0; r1[tid] = s1; __syncthreads();
  for (int o = 64; o > 0; o >>= 1) {
    if (tid < o) { r0[tid] += r0[tid + o]; r1[tid] += r1[tid + o]; }
    __syncthreads();
  }
  if (tid == 0) {
    int b, t;
    if (mode == 0) { b = row; t = it; }
    else { b = row / NTOK; int j = row % NTOK; t = MM + (NTOK - 1 - j); }
    clfpred[((size_t)b * TT + t) * 2 + 0] = r0[0] + b2[0];
    clfpred[((size_t)b * TT + t) * 2 + 1] = r1[0] + b2[1];
  }
}

// ---------------- pipeline glue ----------------
__global__ void k_place_lift(const float* __restrict__ ul, float* __restrict__ u,
                             _Float16* __restrict__ uh) {
  int i = blockIdx.x * blockDim.x + threadIdx.x;
  if (i >= BB * NTOK * ED) return;
  int e = i % ED, row = i / ED;
  int b = row / NTOK, n = row % NTOK;
  float v = ul[i];
  size_t o = ((size_t)b * TT + n) * ED + e;
  u[o] = v; uh[o] = (_Float16)v;
}

__global__ void k_init_active(int* active, int* acnt) {
  int i = blockIdx.x * blockDim.x + threadIdx.x;
  if (i < BB * NTOK) active[i] = i % NTOK;
  if (i < BB) acnt[i] = NTOK;
}

__global__ void k_gather_pairs_init(const _Float16* __restrict__ uh,
                                    _Float16* __restrict__ px, int row0, int nrows) {
  int i = blockIdx.x * blockDim.x + threadIdx.x;
  if (i >= nrows * INPD) return;
  int lrow = i / INPD, e = i % INPD;
  int grow = row0 + lrow;
  int b = grow / 240, p = grow % 240;
  int ii = p / (NTOK - 1), jr = p % (NTOK - 1);
  int jj = jr + (jr >= ii ? 1 : 0);
  int tok = (e < ED) ? ii : jj;
  px[(size_t)lrow * INPD + e] = uh[((size_t)b * TT + tok) * ED + (e & (ED - 1))];
}

__global__ void k_scatter_logits_init(const float* __restrict__ lg, float* __restrict__ lm,
                                      int row0, int nrows) {
  int i = blockIdx.x * blockDim.x + threadIdx.x;
  if (i >= nrows) return;
  int grow = row0 + i;
  int b = grow / 240, p = grow % 240;
  int ii = p / (NTOK - 1), jr = p % (NTOK - 1);
  int jj = jr + (jr >= ii ? 1 : 0);
  lm[(size_t)b * P2T + ii * LL + jj] = lg[i];
}

__global__ void k_gather_pairs_iter(const _Float16* __restrict__ uh,
                                    const int* __restrict__ active,
                                    _Float16* __restrict__ px, int it, int npb) {
  int total = 2 * BB * npb * INPD;
  int i = blockIdx.x * blockDim.x + threadIdx.x;
  if (i >= total) return;
  int row = i / INPD, e = i % INPD;
  int bdir = row >= BB * npb;
  int r2 = bdir ? row - BB * npb : row;
  int b = r2 / npb, q = r2 % npb;
  int nt = NTOK + it - 1;
  int atok = active[b * NTOK + q];
  int first  = bdir ? atok : nt;
  int second = bdir ? nt : atok;
  int tok = (e < ED) ? first : second;
  px[(size_t)row * INPD + e] = uh[((size_t)b * TT + tok) * ED + (e & (ED - 1))];
}

__global__ void k_scatter_logits_iter(const float* __restrict__ lg, float* __restrict__ lm,
                                      const int* __restrict__ active, int it, int npb) {
  int i = blockIdx.x * blockDim.x + threadIdx.x;
  if (i >= 2 * BB * npb) return;
  int bdir = i >= BB * npb;
  int r2 = bdir ? i - BB * npb : i;
  int b = r2 / npb, q = r2 % npb;
  int nt = NTOK + it - 1;
  int atok = active[b * NTOK + q];
  if (!bdir) lm[(size_t)b * P2T + nt * LL + atok] = lg[i];
  else       lm[(size_t)b * P2T + atok * LL + nt] = lg[i];
}

// softmax over 900 logits, entropy accumulation, deterministic "sample" (argmax),
// masking of the chosen pair, active-set compaction.
__global__ void k_softmax_sample(float* __restrict__ lm, float* __restrict__ entropies,
                                 float* __restrict__ log_probs, int* __restrict__ actions,
                                 int* __restrict__ active, int* __restrict__ acnt, int it) {
  const int b = blockIdx.x;
  const int tid = threadIdx.x;
  float* l = lm + (size_t)b * P2T;
  __shared__ float rv[256];
  __shared__ int   ri[256];
  __shared__ float sm, sz;
  __shared__ int   ss0, ss1, ssf;

  float bm = -INFINITY; int bi = 0;
  for (int k = tid; k < P2T; k += 256) { float v = l[k]; if (v > bm) { bm = v; bi = k; } }
  rv[tid] = bm; ri[tid] = bi; __syncthreads();
  for (int o = 128; o > 0; o >>= 1) {
    if (tid < o) {
      if (rv[tid + o] > rv[tid] || (rv[tid + o] == rv[tid] && ri[tid + o] < ri[tid])) {
        rv[tid] = rv[tid + o]; ri[tid] = ri[tid + o];
      }
    }
    __syncthreads();
  }
  if (tid == 0) { sm = rv[0]; ssf = ri[0]; ss0 = ri[0] / LL; ss1 = ri[0] % LL; }
  __syncthreads();
  const float m = sm;

  float z = 0.0f;
  for (int k = tid; k < P2T; k += 256) z += expf(l[k] - m);
  rv[tid] = z; __syncthreads();
  for (int o = 128; o > 0; o >>= 1) { if (tid < o) rv[tid] += rv[tid + o]; __syncthreads(); }
  if (tid == 0) sz = rv[0];
  __syncthreads();
  const float Z = sz;

  const float inv_adj = 1.0f / (1.0f + (float)P2T * SFT_EPS);
  float s = 0.0f;
  for (int k = tid; k < P2T; k += 256) {
    float p  = expf(l[k] - m) / Z;
    float pa = (p + SFT_EPS) * inv_adj;
    s += pa * logf(pa);
  }
  rv[tid] = s; __syncthreads();
  for (int o = 128; o > 0; o >>= 1) { if (tid < o) rv[tid] += rv[tid + o]; __syncthreads(); }

  if (tid == 0) {
    int A = NTOK - it;
    float vn   = SFT_EPS * inv_adj;
    float nopt = (float)(P2T - A * (A - 1));
    float ent  = -(rv[0] - nopt * (vn * logf(vn))) / logf((float)(A * (A - 1)));
    entropies[b] += ent;
    actions[((size_t)b * MM + it) * 2 + 0] = ss0;
    actions[((size_t)b * MM + it) * 2 + 1] = ss1;
    log_probs[(size_t)b * MM + it] = l[ssf] - m - logf(Z);
  }
  __syncthreads();

  for (int k = tid; k < LL; k += 256) {
    l[ss0 * LL + k] = MASKV; l[ss1 * LL + k] = MASKV;
    l[k * LL + ss0] = MASKV; l[k * LL + ss1] = MASKV;
  }
  if (tid == 0) {
    int cnt = acnt[b], outn = 0, tmp[NTOK];
    for (int q = 0; q < cnt; ++q) {
      int t = active[b * NTOK + q];
      if (t != ss0 && t != ss1) tmp[outn++] = t;
    }
    for (int q = 0; q < cnt - 2 && q < outn; ++q) active[b * NTOK + q] = tmp[q];
    active[b * NTOK + cnt - 2] = NTOK + it;
    acnt[b] = cnt - 1;
  }
}

__global__ void k_build_merge_pair(const float* __restrict__ u, const int* __restrict__ actions,
                                   float* __restrict__ pf, _Float16* __restrict__ ph, int it) {
  int i = blockIdx.x * blockDim.x + threadIdx.x;
  if (i >= BB * INPD) return;
  int b = i / INPD, e = i % INPD;
  int s0 = actions[((size_t)b * MM + it) * 2 + 0];
  int s1 = actions[((size_t)b * MM + it) * 2 + 1];
  int tok = (e < ED) ? s0 : s1;
  float v = u[((size_t)b * TT + tok) * ED + (e & (ED - 1))];
  pf[i] = v; ph[i] = (_Float16)v;
}

__global__ void k_osl(const float* __restrict__ pred, const float* __restrict__ pf,
                      float* __restrict__ step_losses, float* __restrict__ rewards, int it) {
  const int b = blockIdx.x;
  const int tid = threadIdx.x;
  float s = 0.0f;
  for (int e = tid; e < INPD; e += 256) {
    float d = pred[(size_t)b * INPD + e] - pf[(size_t)b * INPD + e];
    s += d * d;
  }
  __shared__ float red[256];
  red[tid] = s; __syncthreads();
  for (int o = 128; o > 0; o >>= 1) { if (tid < o) red[tid] += red[tid + o]; __syncthreads(); }
  if (tid == 0) {
    float osl = red[0] / (float)(2 * ED);
    step_losses[b] += osl;
    rewards[(size_t)b * MM + it] = -osl;
  }
}

__global__ void k_append_u(float* __restrict__ u, _Float16* __restrict__ uh,
                           const float* __restrict__ mg, int it) {
  int i = blockIdx.x * blockDim.x + threadIdx.x;
  if (i >= BB * ED) return;
  int b = i / ED, e = i % ED;
  float v = mg[i];
  size_t o = ((size_t)b * TT + NTOK + it) * ED + e;
  u[o] = v; uh[o] = (_Float16)v;
}

__global__ void k_reinf(const float* __restrict__ rewards, const float* __restrict__ log_probs,
                        float* __restrict__ reinf) {
  const int tid = threadIdx.x;
  const int n = BB * MM;
  __shared__ float red[256];
  __shared__ float smean, ssd;
  float s = 0.0f;
  for (int i = tid; i < n; i += 256) s += rewards[i];
  red[tid] = s; __syncthreads();
  for (int o = 128; o > 0; o >>= 1) { if (tid < o) red[tid] += red[tid + o]; __syncthreads(); }
  if (tid == 0) smean = red[0] / (float)n;
  __syncthreads();
  float mean = smean;
  float v = 0.0f;
  for (int i = tid; i < n; i += 256) { float d = rewards[i] - mean; v += d * d; }
  red[tid] = v; __syncthreads();
  for (int o = 128; o > 0; o >>= 1) { if (tid < o) red[tid] += red[tid + o]; __syncthreads(); }
  if (tid == 0) ssd = sqrtf(red[0] / (float)(n - 1)) + STD_EPS;
  __syncthreads();
  float sd = ssd;
  if (tid < BB) {
    float r = 0.0f;
    for (int it = 0; it < MM; ++it)
      r += log_probs[tid * MM + it] * (rewards[tid * MM + it] - mean) / sd;
    reinf[tid] = r;
  }
}

__global__ void k_set_d_active(float* __restrict__ d, const float* __restrict__ u,
                               const int* __restrict__ active, const int* __restrict__ acnt) {
  const int b = blockIdx.x;
  const int tid = threadIdx.x;
  int cnt = acnt[b];
  for (int q = 0; q < cnt; ++q) {
    int t = active[b * NTOK + q];
    for (int e = tid; e < ED; e += 256)
      d[((size_t)b * TT + t) * ED + e] = u[((size_t)b * TT + t) * ED + e];
  }
}

__global__ void k_gather_tok_h(const float* __restrict__ d, int tok, _Float16* __restrict__ mh) {
  int i = blockIdx.x * blockDim.x + threadIdx.x;
  if (i >= BB * ED) return;
  int b = i / ED, e = i % ED;
  mh[i] = (_Float16)d[((size_t)b * TT + tok) * ED + e];
}

__global__ void k_scatter_unmerge(float* __restrict__ d, const float* __restrict__ unmg,
                                  const int* __restrict__ actions, int it) {
  int i = blockIdx.x * blockDim.x + threadIdx.x;
  if (i >= BB * INPD) return;
  int b = i / INPD, e = i % INPD;
  int st = MM - 1 - it;
  int a0 = actions[((size_t)b * MM + st) * 2 + 0];
  int a1 = actions[((size_t)b * MM + st) * 2 + 1];
  int tok = (e < ED) ? a0 : a1;
  d[((size_t)b * TT + tok) * ED + (e & (ED - 1))] = unmg[i];
}

__global__ void k_cvt_dn_h(const float* __restrict__ d, _Float16* __restrict__ dnh) {
  int i = blockIdx.x * blockDim.x + threadIdx.x;
  if (i >= BB * NTOK * ED) return;
  int e = i % ED, row = i / ED;
  int b = row / NTOK, r = row % NTOK;
  dnh[i] = (_Float16)d[((size_t)b * TT + r) * ED + e];
}

// ---------------- host orchestration ----------------
extern "C" void kernel_launch(void* const* d_in, const int* in_sizes, int n_in,
                              void* d_out, int out_size, void* d_ws, size_t ws_size,
                              hipStream_t stream) {
  const float* X        = (const float*)d_in[0];
  const float* lift_w   = (const float*)d_in[1];
  const float* lift_b   = (const float*)d_in[2];
  const float* unlift_w = (const float*)d_in[3];
  const float* unlift_b = (const float*)d_in[4];
  const float* enc_w1   = (const float*)d_in[5];
  const float* enc_b1   = (const float*)d_in[6];
  const float* enc_w2   = (const float*)d_in[7];
  const float* enc_b2   = (const float*)d_in[8];
  const float* enc_w3   = (const float*)d_in[9];
  const float* enc_b3   = (const float*)d_in[10];
  const float* dec_w1   = (const float*)d_in[11];
  const float* dec_b1   = (const float*)d_in[12];
  const float* dec_w2   = (const float*)d_in[13];
  const float* dec_b2   = (const float*)d_in[14];
  const float* dec_w3   = (const float*)d_in[15];
  const float* dec_b3   = (const float*)d_in[16];
  const float* clf_w1   = (const float*)d_in[17];
  const float* clf_b1   = (const float*)d_in[18];
  const float* clf_w2   = (const float*)d_in[19];
  const float* clf_b2   = (const float*)d_in[20];
  const float* clf_w3   = (const float*)d_in[21];
  const float* clf_b3   = (const float*)d_in[22];
  const float* pol_w1   = (const float*)d_in[23];
  const float* pol_b1   = (const float*)d_in[24];
  const float* pol_w2   = (const float*)d_in[25];
  const float* pol_b2   = (const float*)d_in[26];
  const float* pol_w3   = (const float*)d_in[27];
  const float* pol_b3   = (const float*)d_in[28];

  float* out = (float*)d_out;
  const size_t o_recon = 0;
  const size_t o_u     = o_recon + (size_t)BB * NTOK * INPD;
  const size_t o_d     = o_u + (size_t)BB * TT * ED;
  const size_t o_sl    = o_d + (size_t)BB * TT * ED;
  const size_t o_ent   = o_sl + BB;
  const size_t o_clf   = o_ent + BB;
  const size_t o_lbl   = o_clf + (size_t)BB * TT * 2;
  const size_t o_reinf = o_lbl + (size_t)BB * TT;

  char* base = (char*)d_ws;
  size_t off = 0;
  auto alloc = [&](size_t bytes) -> void* {
    void* p = base + off;
    off = (off + bytes + 255) & ~(size_t)255;
    return p;
  };

  // transposed f16 weights
  _Float16* lift_wt = (_Float16*)alloc((size_t)INPD * ED * 2);
  _Float16* unlif_t = (_Float16*)alloc((size_t)ED * INPD * 2);
  _Float16* enc1t   = (_Float16*)alloc((size_t)INPD * ED * 2);
  _Float16* enc2t   = (_Float16*)alloc((size_t)ED * ED * 2);
  _Float16* enc3t   = (_Float16*)alloc((size_t)ED * ED * 2);
  _Float16* dec1t   = (_Float16*)alloc((size_t)ED * ED * 2);
  _Float16* dec2t   = (_Float16*)alloc((size_t)ED * ED * 2);
  _Float16* dec3t   = (_Float16*)alloc((size_t)ED * INPD * 2);
  _Float16* clf1t   = (_Float16*)alloc((size_t)ED * ED * 2);
  _Float16* clf2t   = (_Float16*)alloc((size_t)ED * ED * 2);
  _Float16* pol1t   = (_Float16*)alloc((size_t)INPD * ED * 2);
  _Float16* pol2t   = (_Float16*)alloc((size_t)ED * ED * 2);

  _Float16* xh     = (_Float16*)alloc((size_t)BB * NTOK * INPD * 2);
  float*    ulift  = (float*)alloc((size_t)BB * NTOK * ED * 4);
  float*    u      = (float*)alloc((size_t)BB * TT * ED * 4);
  _Float16* uh     = (_Float16*)alloc((size_t)BB * TT * ED * 2);
  float*    lm     = (float*)alloc((size_t)BB * P2T * 4);

  const int CH = 3840;                       // policy chunk rows
  _Float16* px   = (_Float16*)alloc((size_t)CH * INPD * 2);
  _Float16* h1h  = (_Float16*)alloc((size_t)CH * ED * 2);
  _Float16* h2h  = (_Float16*)alloc((size_t)CH * ED * 2);
  float*    lgts = (float*)alloc((size_t)CH * 4);

  int*   active  = (int*)alloc((size_t)BB * NTOK * 4);
  int*   acnt    = (int*)alloc((size_t)BB * 4);
  int*   actions = (int*)alloc((size_t)BB * MM * 2 * 4);
  float* lprob   = (float*)alloc((size_t)BB * MM * 4);
  float* rew     = (float*)alloc((size_t)BB * MM * 4);
  float* sloss   = (float*)alloc((size_t)BB * 4);
  float* entrop  = (float*)alloc((size_t)BB * 4);

  float*    pairf = (float*)alloc((size_t)BB * INPD * 4);
  _Float16* pairh = (_Float16*)alloc((size_t)BB * INPD * 2);
  _Float16* e1h   = (_Float16*)alloc((size_t)BB * ED * 2);
  _Float16* e2h   = (_Float16*)alloc((size_t)BB * ED * 2);
  float*    mg    = (float*)alloc((size_t)BB * ED * 4);
  _Float16* mgh   = (_Float16*)alloc((size_t)BB * ED * 2);
  _Float16* pd1h  = (_Float16*)alloc((size_t)BB * ED * 2);
  _Float16* pd2h  = (_Float16*)alloc((size_t)BB * ED * 2);
  float*    pred  = (float*)alloc((size_t)BB * INPD * 4);

  float*    d    = (float*)alloc((size_t)BB * TT * ED * 4);
  _Float16* mgdh = (_Float16*)alloc((size_t)BB * ED * 2);
  float*    unmg = (float*)alloc((size_t)BB * INPD * 4);
  _Float16* c1h  = (_Float16*)alloc((size_t)BB * NTOK * ED * 2);  // big enough for final clf
  _Float16* c2h  = (_Float16*)alloc((size_t)BB * NTOK * ED * 2);
  _Float16* dnh  = (_Float16*)alloc((size_t)BB * NTOK * ED * 2);

  auto cdiv = [](int a, int b) { return (a + b - 1) / b; };
  auto cvt = [&](const float* s, _Float16* dst, int n) {
    k_cvt_f16<<<cdiv(n, 256), 256, 0, stream>>>(s, dst, n);
  };
  auto ctr = [&](const float* W, _Float16* Wt, int K, int Nc) {
    k_cvt_tr_f16<<<cdiv(K * Nc, 256), 256, 0, stream>>>(W, Wt, K, Nc);
  };
  auto gemm = [&](const _Float16* A, const _Float16* Bt, const float* bias,
                  float* C, _Float16* Ch, int Mr, int K, int Nc, int relu) {
    dim3 g(Mr / TBM, Nc / TBN);
    k_gemm_f16<<<g, 256, 0, stream>>>(A, Bt, bias, C, Ch, K, Nc, relu);
  };

  // ---- weight prep ----
  ctr(lift_w, lift_wt, INPD, ED);
  ctr(unlift_w, unlif_t, ED, INPD);
  ctr(enc_w1, enc1t, INPD, ED);  ctr(enc_w2, enc2t, ED, ED);  ctr(enc_w3, enc3t, ED, ED);
  ctr(dec_w1, dec1t, ED, ED);    ctr(dec_w2, dec2t, ED, ED);  ctr(dec_w3, dec3t, ED, INPD);
  ctr(clf_w1, clf1t, ED, ED);    ctr(clf_w2, clf2t, ED, ED);
  ctr(pol_w1, pol1t, INPD, ED);  ctr(pol_w2, pol2t, ED, ED);

  // ---- state init ----
  k_fill<<<cdiv(BB, 256), 256, 0, stream>>>(sloss, 0.0f, BB);
  k_fill<<<cdiv(BB, 256), 256, 0, stream>>>(entrop, 0.0f, BB);
  k_fill<<<cdiv(BB * P2T, 256), 256, 0, stream>>>(lm, MASKV, BB * P2T);
  k_init_active<<<cdiv(BB * NTOK, 256), 256, 0, stream>>>(active, acnt);

  // ---- lift ----
  cvt(X, xh, BB * NTOK * INPD);
  gemm(xh, lift_wt, lift_b, ulift, nullptr, BB * NTOK, INPD, ED, 0);
  k_place_lift<<<cdiv(BB * NTOK * ED, 256), 256, 0, stream>>>(ulift, u, uh);

  // ---- initial policy over all (i,j) pairs, chunked ----
  for (int c = 0; c < (BB * 240) / CH; ++c) {
    int row0 = c * CH;
    k_gather_pairs_init<<<cdiv(CH * INPD, 256), 256, 0, stream>>>(uh, px, row0, CH);
    gemm(px, pol1t, pol_b1, nullptr, h1h, CH, INPD, ED, 1);
    gemm(h1h, pol2t, pol_b2, nullptr, h2h, CH, ED, ED, 1);
    k_rowdot1<<<CH, 128, 0, stream>>>(h2h, pol_w3, pol_b3, lgts, ED);
    k_scatter_logits_init<<<cdiv(CH, 256), 256, 0, stream>>>(lgts, lm, row0, CH);
  }

  // ---- merge loop ----
  for (int it = 0; it < MM; ++it) {
    if (it > 0) {
      int npb = NTOK - it - 1;
      int rows = 2 * BB * npb;
      k_gather_pairs_iter<<<cdiv(rows * INPD, 256), 256, 0, stream>>>(uh, active, px, it, npb);
      gemm(px, pol1t, pol_b1, nullptr, h1h, rows, INPD, ED, 1);
      gemm(h1h, pol2t, pol_b2, nullptr, h2h, rows, ED, ED, 1);
      k_rowdot1<<<rows, 128, 0, stream>>>(h2h, pol_w3, pol_b3, lgts, ED);
      k_scatter_logits_iter<<<cdiv(rows, 256), 256, 0, stream>>>(lgts, lm, active, it, npb);
    }
    k_softmax_sample<<<BB, 256, 0, stream>>>(lm, entrop, lprob, actions, active, acnt, it);
    k_build_merge_pair<<<cdiv(BB * INPD, 256), 256, 0, stream>>>(u, actions, pairf, pairh, it);
    // enc
    gemm(pairh, enc1t, enc_b1, nullptr, e1h, BB, INPD, ED, 1);
    gemm(e1h, enc2t, enc_b2, nullptr, e2h, BB, ED, ED, 1);
    gemm(e2h, enc3t, enc_b3, mg, mgh, BB, ED, ED, 0);
    // dec(enc) for reconstruction loss
    gemm(mgh, dec1t, dec_b1, nullptr, pd1h, BB, ED, ED, 1);
    gemm(pd1h, dec2t, dec_b2, nullptr, pd2h, BB, ED, ED, 1);
    gemm(pd2h, dec3t, dec_b3, pred, nullptr, BB, ED, INPD, 0);
    k_osl<<<BB, 256, 0, stream>>>(pred, pairf, sloss, rew, it);
    k_append_u<<<cdiv(BB * ED, 256), 256, 0, stream>>>(u, uh, mg, it);
  }

  // ---- reinforce ----
  k_reinf<<<1, 256, 0, stream>>>(rew, lprob, out + o_reinf);

  // ---- build d and unmerge ----
  k_fill<<<cdiv(BB * TT * ED, 256), 256, 0, stream>>>(d, 0.0f, BB * TT * ED);
  k_set_d_active<<<BB, 256, 0, stream>>>(d, u, active, acnt);

  for (int it = 0; it < MM; ++it) {
    int tok = TT - 1 - it;
    k_gather_tok_h<<<cdiv(BB * ED, 256), 256, 0, stream>>>(d, tok, mgdh);
    // dec to unmerge
    gemm(mgdh, dec1t, dec_b1, nullptr, pd1h, BB, ED, ED, 1);
    gemm(pd1h, dec2t, dec_b2, nullptr, pd2h, BB, ED, ED, 1);
    gemm(pd2h, dec3t, dec_b3, unmg, nullptr, BB, ED, INPD, 0);
    k_scatter_unmerge<<<cdiv(BB * INPD, 256), 256, 0, stream>>>(d, unmg, actions, it);
    // clf on merged token
    gemm(mgdh, clf1t, clf_b1, nullptr, c1h, BB, ED, ED, 1);
    gemm(c1h, clf2t, clf_b2, nullptr, c2h, BB, ED, ED, 1);
    k_clf_head<<<BB, 128, 0, stream>>>(c2h, clf_w3, clf_b3, out + o_clf, ED, 0, it);
  }

  // ---- final clf on d[:, :N] (reversed token order) ----
  k_cvt_dn_h<<<cdiv(BB * NTOK * ED, 256), 256, 0, stream>>>(d, dnh);
  gemm(dnh, clf1t, clf_b1, nullptr, c1h, BB * NTOK, ED, ED, 1);
  gemm(c1h, clf2t, clf_b2, nullptr, c2h, BB * NTOK, ED, ED, 1);
  k_clf_head<<<BB * NTOK, 128, 0, stream>>>(c2h, clf_w3, clf_b3, out + o_clf, ED, 1, 0);

  // ---- recon = d[:, :N] @ unlift + b ----
  gemm(dnh, unlif_t, unlift_b, out + o_recon, nullptr, BB * NTOK, ED, INPD, 0);

  // ---- remaining outputs ----
  k_copy<<<cdiv(BB * TT * ED, 256), 256, 0, stream>>>(out + o_u, u, BB * TT * ED);
  k_copy<<<cdiv(BB * TT * ED, 256), 256, 0, stream>>>(out + o_d, d, BB * TT * ED);
  k_scale_copy<<<1, 256, 0, stream>>>(out + o_sl, sloss, 1.0f / (float)MM, BB);
  k_scale_copy<<<1, 256, 0, stream>>>(out + o_ent, entrop, 1.0f / (float)MM, BB);
  k_labels<<<cdiv(BB * TT, 256), 256, 0, stream>>>(out + o_lbl);
  (void)in_sizes; (void)n_in; (void)out_size; (void)ws_size;
}
